// BaselineGNN_9277129360054
// MI455X (gfx1250) — matrix operand order repaired and checked by reading, not compiled, not run on hardware
//
#include <hip/hip_runtime.h>
#include <hip/hip_bf16.h>
#include <stdint.h>

// ---------------- problem constants (from the reference) ----------------
#define IN_CH  256
#define HEADS  4
#define CPH    64
#define HC     256   // HEADS*CPH
#define HIDW   64
#define OUTC   40
#define OUTP   48    // OUTC padded to a multiple of 16 for WMMA
#define NEG_SLOPE 0.2f

typedef __attribute__((ext_vector_type(16))) __bf16 bf16x16;
typedef __attribute__((ext_vector_type(8)))  __bf16 bf16x8;
typedef __attribute__((ext_vector_type(8)))  float  floatx8;

// fp32 -> bf16, round-to-nearest-even
__device__ __forceinline__ __bf16 f2bf(float f) {
    unsigned u = __float_as_uint(f);
    unsigned r = (u + 0x7FFFu + ((u >> 16) & 1u)) >> 16;
    unsigned short s = (unsigned short)r;
    return __builtin_bit_cast(__bf16, s);
}

// order-preserving float <-> uint mapping (for atomic float max via atomicMax(u32))
__device__ __forceinline__ unsigned fmap(float f) {
    unsigned u = __float_as_uint(f);
    return (u & 0x80000000u) ? ~u : (u | 0x80000000u);
}
__device__ __forceinline__ float funmap(unsigned m) {
    return __uint_as_float((m & 0x80000000u) ? (m & 0x7FFFFFFFu) : ~m);
}

__device__ __forceinline__ float lrelu(float v) { return v > 0.f ? v : NEG_SLOPE * v; }

// ---------------- elementwise converts ----------------
__global__ void k_cvt_bf16(const float* __restrict__ in, __bf16* __restrict__ out, long n) {
    long i = (long)blockIdx.x * blockDim.x + threadIdx.x;
    if (i < n) out[i] = f2bf(in[i]);
}

// BT[col*K + k] = (col < ncsrc) ? in[k*ncsrc + col] : 0   (convert + transpose + pad)
__global__ void k_cvt_transpose(const float* __restrict__ in, __bf16* __restrict__ out,
                                int K, int ncsrc, int ncdst) {
    int t = blockIdx.x * blockDim.x + threadIdx.x;
    if (t >= K * ncdst) return;
    int col = t / K, k = t - col * K;
    float v = (col < ncsrc) ? in[(long)k * ncsrc + col] : 0.f;
    out[(long)col * K + k] = f2bf(v);
}

// ---------------- WMMA GEMM: C[M,ncol] = A[M,K] (bf16) x BT[ncol,K]^T (bf16) ----------------
// one wave per 16x16 tile; grid = (Mtiles, NcolTiles)
template <bool RELU, bool BF16OUT>
__global__ void k_wmma_gemm(const __bf16* __restrict__ A, const __bf16* __restrict__ BT,
                            const float* __restrict__ bias,
                            float* __restrict__ Cf, __bf16* __restrict__ Cb,
                            int Mvalid, int K, int ldc, int ncol_valid) {
    const int trow = blockIdx.x;
    const int tcol = blockIdx.y;
    const int lane = threadIdx.x;            // wave32
    const int r    = lane & 15;
    const int hf   = lane >> 4;              // half-wave id

    // A: 16x32 tile, k order chosen here (not the ISA table order) -> lane holds K {hf*16 .. hf*16+15} contiguous
    const size_t abase = (size_t)(trow * 16 + r) * K + hf * 16;
    // B: 32x16 tile, same k order as A -> lane holds K {hf*16 .. hf*16+15} contiguous
    const size_t bbase = (size_t)(tcol * 16 + r) * K + hf * 16;

    floatx8 acc = {};
    for (int k = 0; k < K; k += 32) {
        bf16x8 a0 = *(const bf16x8*)(A + abase + k);
        bf16x8 a1 = *(const bf16x8*)(A + abase + k + 8);
        bf16x16 bv = *(const bf16x16*)(BT + bbase + k);
        bf16x16 av;
#pragma unroll
        for (int i = 0; i < 8; ++i) { av[i] = a0[i]; av[i + 8] = a1[i]; }
        acc = __builtin_amdgcn_wmma_f32_16x16x32_bf16(
            /*neg_a=*/false, av, /*neg_b=*/false, bv,
            /*c_mod=*/(short)0, acc, /*reuse_a=*/false, /*reuse_b=*/false);
    }

    const int col  = tcol * 16 + r;
    if (col >= ncol_valid) return;
    const float bv = bias ? bias[col] : 0.f;
    const int row0 = trow * 16 + hf * 8;     // C layout: VGPR v -> row row0+v
#pragma unroll
    for (int v = 0; v < 8; ++v) {
        int row = row0 + v;
        if (row >= Mvalid) continue;
        float val = acc[v] + bv;
        if (RELU) val = fmaxf(val, 0.f);
        size_t o = (size_t)row * ldc + col;
        if (BF16OUT) Cb[o] = f2bf(val); else Cf[o] = val;
    }
}

// ---------------- per-node attention dots: a_src/a_dst [N,4] ----------------
__global__ void k_attn(const float* __restrict__ h, const float* __restrict__ att_src,
                       const float* __restrict__ att_dst,
                       float* __restrict__ a_src, float* __restrict__ a_dst, int N) {
    int t = blockIdx.x * blockDim.x + threadIdx.x;
    if (t >= N * HEADS) return;
    int n = t >> 2, hd = t & 3;
    const float4* hp  = (const float4*)(h + (size_t)n * HC + hd * CPH);
    const float4* asp = (const float4*)(att_src + hd * CPH);
    const float4* adp = (const float4*)(att_dst + hd * CPH);
    float ss = 0.f, sd = 0.f;
#pragma unroll
    for (int i = 0; i < CPH / 4; ++i) {
        float4 hv = hp[i], s4 = asp[i], d4 = adp[i];
        ss += hv.x * s4.x + hv.y * s4.y + hv.z * s4.z + hv.w * s4.w;
        sd += hv.x * d4.x + hv.y * d4.y + hv.z * d4.z + hv.w * d4.w;
    }
    a_src[t] = ss; a_dst[t] = sd;
}

// ---------------- edge logits + segment max (E2 = E + N self-loops) ----------------
__global__ void k_edge_logits(const int* __restrict__ ei, const float* __restrict__ a_src,
                              const float* __restrict__ a_dst, float* __restrict__ ebuf,
                              unsigned* __restrict__ m_u, int E, int E2) {
    int e = blockIdx.x * blockDim.x + threadIdx.x;
    if (e >= E2) return;
    int s, d;
    if (e < E) { s = ei[e]; d = ei[E + e]; } else { s = e - E; d = s; }
    float4 as = *(const float4*)(a_src + (size_t)s * 4);
    float4 ad = *(const float4*)(a_dst + (size_t)d * 4);
    float4 ev;
    ev.x = lrelu(as.x + ad.x); ev.y = lrelu(as.y + ad.y);
    ev.z = lrelu(as.z + ad.z); ev.w = lrelu(as.w + ad.w);
    *(float4*)(ebuf + (size_t)e * 4) = ev;
    unsigned* mp = m_u + (size_t)d * 4;
    atomicMax(mp + 0, fmap(ev.x)); atomicMax(mp + 1, fmap(ev.y));
    atomicMax(mp + 2, fmap(ev.z)); atomicMax(mp + 3, fmap(ev.w));
}

// ---------------- exp(e - max) + segment sum ----------------
__global__ void k_edge_exp(const int* __restrict__ ei, float* __restrict__ ebuf,
                           const unsigned* __restrict__ m_u, float* __restrict__ denom,
                           int E, int E2) {
    int e = blockIdx.x * blockDim.x + threadIdx.x;
    if (e >= E2) return;
    int d = (e < E) ? ei[E + e] : (e - E);
    const unsigned* mp = m_u + (size_t)d * 4;
    float4 ev = *(const float4*)(ebuf + (size_t)e * 4);
    ev.x = expf(ev.x - funmap(mp[0])); ev.y = expf(ev.y - funmap(mp[1]));
    ev.z = expf(ev.z - funmap(mp[2])); ev.w = expf(ev.w - funmap(mp[3]));
    *(float4*)(ebuf + (size_t)e * 4) = ev;
    float* dp = denom + (size_t)d * 4;
    atomicAdd(dp + 0, ev.x); atomicAdd(dp + 1, ev.y);
    atomicAdd(dp + 2, ev.z); atomicAdd(dp + 3, ev.w);
}

__global__ void k_invdenom(const float* __restrict__ denom, float* __restrict__ invden, int n) {
    int t = blockIdx.x * blockDim.x + threadIdx.x;
    if (t < n) invden[t] = 1.0f / denom[t];   // every segment has a self-loop -> denom > 0
}

// ---------------- weighted scatter-aggregate: agg[d] += alpha * h[s] ----------------
// 64 lanes per edge, 4 channels (float4) per lane
__global__ void k_aggregate(const int* __restrict__ ei, const float* __restrict__ ebuf,
                            const float* __restrict__ invden, const float* __restrict__ h,
                            float* __restrict__ agg, int E, int E2) {
    long t = (long)blockIdx.x * blockDim.x + threadIdx.x;
    long e = t >> 6;
    if (e >= E2) return;
    int sub = (int)(t & 63);
    int c   = sub * 4;
    int hd  = sub >> 4;
    int s, d;
    if (e < E) { s = ei[e]; d = ei[E + e]; } else { s = (int)(e - E); d = s; }
    float alpha = ebuf[e * 4 + hd] * invden[(size_t)d * 4 + hd];
    float4 hv = *(const float4*)(h + (size_t)s * HC + c);
    float* o = agg + (size_t)d * HC + c;
    atomicAdd(o + 0, alpha * hv.x);
    atomicAdd(o + 1, alpha * hv.y);
    atomicAdd(o + 2, alpha * hv.z);
    atomicAdd(o + 3, alpha * hv.w);
}

// ---------------- bias_conv + relu + bf16 convert (input to MLP GEMM) ----------------
__global__ void k_finalize(const float* __restrict__ agg, const float* __restrict__ bias_conv,
                           __bf16* __restrict__ out, long n) {
    long i = (long)blockIdx.x * blockDim.x + threadIdx.x;
    if (i >= n) return;
    int c = (int)(i & (HC - 1));
    out[i] = f2bf(fmaxf(agg[i] + bias_conv[c], 0.f));
}

// ---------------------------------------------------------------------------
extern "C" void kernel_launch(void* const* d_in, const int* in_sizes, int n_in,
                              void* d_out, int out_size, void* d_ws, size_t ws_size,
                              hipStream_t stream) {
    const float* x         = (const float*)d_in[0];
    const int*   ei        = (const int*)d_in[1];   // jnp default: int32, [2,E] row-major
    const float* W         = (const float*)d_in[2];
    const float* att_src   = (const float*)d_in[3];
    const float* att_dst   = (const float*)d_in[4];
    const float* bias_conv = (const float*)d_in[5];
    const float* W1        = (const float*)d_in[6];
    const float* b1        = (const float*)d_in[7];
    const float* W2        = (const float*)d_in[8];
    const float* b2        = (const float*)d_in[9];

    const int N    = in_sizes[0] / IN_CH;
    const int E    = in_sizes[1] / 2;
    const int E2   = E + N;                       // with self-loops
    const int Npad = ((N + 15) / 16) * 16;
    const int mt   = Npad / 16;                   // M tiles

    // ---- workspace carve-out (256B aligned slices) ----
    char* p = (char*)d_ws;
    auto alloc = [&](size_t bytes) -> char* {
        char* r = p; p += (bytes + 255) & ~(size_t)255; return r;
    };
    __bf16* x_bf  = (__bf16*)alloc((size_t)Npad * HC * 2);   // reused as agg_bf after GEMM1
    __bf16* WT    = (__bf16*)alloc((size_t)IN_CH * HC * 2);
    __bf16* W1T   = (__bf16*)alloc((size_t)HC * HIDW * 2);
    __bf16* W2T   = (__bf16*)alloc((size_t)HIDW * OUTP * 2);
    float*  h     = (float*) alloc((size_t)Npad * HC * 4);
    float*  a_s   = (float*) alloc((size_t)N * HEADS * 4);
    float*  a_d   = (float*) alloc((size_t)N * HEADS * 4);
    float*  ebuf  = (float*) alloc((size_t)E2 * HEADS * 4);
    unsigned* m_u = (unsigned*)alloc((size_t)N * HEADS * 4);
    float*  denom = (float*) alloc((size_t)N * HEADS * 4);
    float*  invd  = (float*) alloc((size_t)N * HEADS * 4);
    float*  agg   = (float*) alloc((size_t)Npad * HC * 4);
    __bf16* z_bf  = (__bf16*)alloc((size_t)Npad * HIDW * 2);
    __bf16* agg_bf = x_bf;                                   // alias: x_bf dead after GEMM1

    auto cdiv = [](long a, long b) { return (unsigned)((a + b - 1) / b); };

    // ---- per-call init (graph-capture safe) ----
    hipMemsetAsync(m_u,   0, (size_t)N * HEADS * 4, stream); // 0 maps below any real float
    hipMemsetAsync(denom, 0, (size_t)N * HEADS * 4, stream);
    hipMemsetAsync(agg,   0, (size_t)Npad * HC * 4, stream);

    // ---- converts ----
    k_cvt_bf16<<<cdiv((long)N * HC, 256), 256, 0, stream>>>(x, x_bf, (long)N * HC);
    k_cvt_transpose<<<cdiv((long)IN_CH * HC, 256), 256, 0, stream>>>(W,  WT,  IN_CH, HC,  HC);
    k_cvt_transpose<<<cdiv((long)HC * HIDW, 256), 256, 0, stream>>>(W1, W1T, HC,   HIDW, HIDW);
    k_cvt_transpose<<<cdiv((long)HIDW * OUTP, 256), 256, 0, stream>>>(W2, W2T, HIDW, OUTC, OUTP);

    // ---- GEMM1: h = x @ W  (no bias/relu) ----
    k_wmma_gemm<false, false><<<dim3(mt, HC / 16), 32, 0, stream>>>(
        x_bf, WT, nullptr, h, nullptr, N, IN_CH, HC, HC);

    // ---- attention dots ----
    k_attn<<<cdiv((long)N * HEADS, 256), 256, 0, stream>>>(h, att_src, att_dst, a_s, a_d, N);

    // ---- edge softmax ----
    k_edge_logits<<<cdiv(E2, 256), 256, 0, stream>>>(ei, a_s, a_d, ebuf, m_u, E, E2);
    k_edge_exp   <<<cdiv(E2, 256), 256, 0, stream>>>(ei, ebuf, m_u, denom, E, E2);
    k_invdenom   <<<cdiv((long)N * HEADS, 256), 256, 0, stream>>>(denom, invd, N * HEADS);

    // ---- scatter-aggregate (L2-resident atomics) ----
    k_aggregate<<<cdiv((long)E2 * 64, 256), 256, 0, stream>>>(ei, ebuf, invd, h, agg, E, E2);

    // ---- conv bias + relu -> bf16 ----
    k_finalize<<<cdiv((long)N * HC, 256), 256, 0, stream>>>(agg, bias_conv, agg_bf, (long)N * HC);

    // ---- GEMM2: z = relu(agg @ W1 + b1), bf16 out ----
    k_wmma_gemm<true, true><<<dim3(mt, HIDW / 16), 32, 0, stream>>>(
        agg_bf, W1T, b1, nullptr, z_bf, N, HC, HIDW, HIDW);

    // ---- GEMM3: out = z @ W2 + b2 (40 cols, padded to 48 tiles) ----
    k_wmma_gemm<false, false><<<dim3(mt, OUTP / 16), 32, 0, stream>>>(
        z_bf, W2T, b2, (float*)d_out, nullptr, N, HIDW, OUTC, OUTC);
}
